// RecurrentGNN_13743895347605
// MI455X (gfx1250) — matrix operand order repaired and checked by reading, not compiled, not run on hardware
//
#include <hip/hip_runtime.h>
#include <math.h>

// ---------------------------------------------------------------------------
// MI455X / gfx1250, wave32. Matrix ops via v_wmma_f32_16x16x32_bf16.
// Exploits H=C=0 in GConvLSTM: all Wh GEMMs collapse to bias adds, forget
// gate is dead (F*C == 0). Dense math = bf16 WMMA w/ f32 accum; sparse
// Laplacian applies = edge-parallel gather + f32 atomic scatter (the true
// bottleneck at ~4 GB of HBM traffic; 192MB L2 absorbs GEMM operand reuse).
// ---------------------------------------------------------------------------

typedef __attribute__((ext_vector_type(16))) __bf16 v16bf;
typedef __attribute__((ext_vector_type(8)))  __bf16 v8bf;
typedef __attribute__((ext_vector_type(8)))  float  v8f;

#define LRELU_SLOPE 0.1f

__device__ __forceinline__ v16bf ld_frag(const __bf16* p0, const __bf16* p1) {
  v8bf lo = *(const v8bf*)p0;   // global_load_b128
  v8bf hi = *(const v8bf*)p1;   // global_load_b128
  v16bf r;
#pragma unroll
  for (int i = 0; i < 8; ++i) { r[i] = lo[i]; r[i + 8] = hi[i]; }
  return r;
}

__device__ __forceinline__ float sigmoidf_(float x) { return 1.0f / (1.0f + __expf(-x)); }

// ------------------------- graph preprocessing ----------------------------
__global__ void k_zero_f32(float* p, long long n) {
  long long i = blockIdx.x * (long long)blockDim.x + threadIdx.x;
  if (i < n) p[i] = 0.0f;
}

__global__ void k_degree(const int* __restrict__ src, float* deg, int E) {
  int e = blockIdx.x * blockDim.x + threadIdx.x;
  if (e < E) atomicAdd(&deg[src[e]], 1.0f);
}

__global__ void k_dis(float* deg, int N) {
  int i = blockIdx.x * blockDim.x + threadIdx.x;
  if (i < N) {
    float d = deg[i];
    deg[i] = d > 0.0f ? rsqrtf(fmaxf(d, 1.0f)) : 0.0f;
  }
}

__global__ void k_edge_w(const int* __restrict__ src, const int* __restrict__ dst,
                         const float* __restrict__ dis, float* w, int E) {
  int e = blockIdx.x * blockDim.x + threadIdx.x;
  if (e < E) w[e] = -dis[src[e]] * dis[dst[e]];
}

// ------------------------------ sparse ops --------------------------------
__global__ void k_neg_copy(const float* __restrict__ x, float* y, long long n) {
  long long i = blockIdx.x * (long long)blockDim.x + threadIdx.x;
  if (i < n) y[i] = -x[i];
}

// Y[dst] += scale * w[e] * X[src]   (edge-parallel, feature-coalesced)
__global__ void k_spmm(const float* __restrict__ X, const int* __restrict__ src,
                       const int* __restrict__ dst, const float* __restrict__ w,
                       float* Y, int F, float scale, long long total) {
  long long i = blockIdx.x * (long long)blockDim.x + threadIdx.x;
  if (i >= total) return;
  int e = (int)(i / F);
  int f = (int)(i - (long long)e * F);
  float v = scale * w[e] * X[(size_t)src[e] * F + f];
  atomicAdd(&Y[(size_t)dst[e] * F + f], v);
}

// ------------------------------- packing ----------------------------------
// A_cat[n][0:F)=T0, [F:2F)=T1, [2F:3F)=T2, zero-pad to Kpad (mult of 32)
__global__ void k_pack3_bf16(const float* __restrict__ t0, const float* __restrict__ t1,
                             const float* __restrict__ t2, __bf16* A,
                             int F, int Kpad, long long total) {
  long long i = blockIdx.x * (long long)blockDim.x + threadIdx.x;
  if (i >= total) return;
  long long n = i / Kpad;
  int c = (int)(i - n * Kpad);
  float v = 0.0f;
  if (c < F)          v = t0[n * F + c];
  else if (c < 2 * F) v = t1[n * F + (c - F)];
  else if (c < 3 * F) v = t2[n * F + (c - 2 * F)];
  A[i] = (__bf16)v;
}

__global__ void k_pack_bf16(const float* __restrict__ x, __bf16* A, long long n) {
  long long i = blockIdx.x * (long long)blockDim.x + threadIdx.x;
  if (i < n) A[i] = (__bf16)x[i];
}

// ----------------------------- weight prep --------------------------------
// Wx: f32 [4][3][F][Fout].  Build per-gate (i,c,o -> Wx[0],Wx[2],Wx[3])
// transposed bf16 weights Wt[o][k*F+f], zero-padded K -> Kpad.
__global__ void k_prep_cheb_w(const float* __restrict__ Wx, __bf16* Wti, __bf16* Wtc,
                              __bf16* Wto, int F, int Fout, int Kpad, int total) {
  int i = blockIdx.x * blockDim.x + threadIdx.x;
  if (i >= total) return;
  int g = i / (Fout * Kpad);
  int rem = i - g * (Fout * Kpad);
  int o = rem / Kpad, c = rem - o * Kpad;
  int gate = (g == 0) ? 0 : (g + 1);   // 0,2,3
  float v = 0.0f;
  if (c < 3 * F) {
    int k = c / F, f = c - k * F;
    v = Wx[(((size_t)gate * 3 + k) * F + f) * Fout + o];
  }
  __bf16* W = (g == 0) ? Wti : (g == 1) ? Wtc : Wto;
  W[(size_t)o * Kpad + c] = (__bf16)v;
}

// bias_g[o] = bx[g]+bh[g]+b[g]  (bh because cheb(H=0) == bh); wc2 = wc[2]
__global__ void k_prep_bias(const float* __restrict__ bx, const float* __restrict__ bh,
                            const float* __restrict__ b, const float* __restrict__ wc,
                            float* bi, float* bc, float* bo, float* wc2, int Fout) {
  int o = blockIdx.x * blockDim.x + threadIdx.x;
  if (o >= Fout) return;
  bi[o] = bx[0 * Fout + o] + bh[0 * Fout + o] + b[0 * Fout + o];
  bc[o] = bx[2 * Fout + o] + bh[2 * Fout + o] + b[2 * Fout + o];
  bo[o] = bx[3 * Fout + o] + bh[3 * Fout + o] + b[3 * Fout + o];
  wc2[o] = wc[2 * Fout + o];
}

__global__ void k_prep_lin_w(const float* __restrict__ W, __bf16* Wt,
                             int Kd, int Fout, int total) {
  int i = blockIdx.x * blockDim.x + threadIdx.x;
  if (i >= total) return;
  int o = i / Kd, k = i - o * Kd;
  Wt[(size_t)o * Kd + k] = (__bf16)W[(size_t)k * Fout + o];
}

// --------------------- fused cheb + LSTM-cell WMMA GEMM -------------------
// One wave per 16x16 output tile; 3 accumulators (gates i,c,o) share the A
// fragment.  A: row-major bf16 N x Kpad; Wt*: col-major bf16 Fout x Kpad.
// A/B frags use contiguous k order (lane half h holds k 16h..16h+15); C/D f32 layout per ISA 7.12.2.
__global__ void __launch_bounds__(128)
k_cheb_lstm_wmma(const __bf16* __restrict__ A, int Kpad,
                 const __bf16* __restrict__ Wti, const __bf16* __restrict__ Wtc,
                 const __bf16* __restrict__ Wto,
                 const float* __restrict__ bi, const float* __restrict__ bc,
                 const float* __restrict__ bo, const float* __restrict__ wc2,
                 float* out, int mTiles, int nTiles, int Fout) {
  int wave = threadIdx.x >> 5;
  int lane = threadIdx.x & 31;
  int tile = blockIdx.x * 4 + wave;
  if (tile >= mTiles * nTiles) return;          // wave-uniform: EXEC stays full
  int nTile = tile % nTiles;
  int mTile = tile / nTiles;
  int half = lane >> 4;
  int l16 = lane & 15;

  const __bf16* arow = A + (size_t)(mTile * 16 + l16) * Kpad;
  size_t wrow = (size_t)(nTile * 16 + l16) * Kpad;

  v8f accI = {}; v8f accC = {}; v8f accO = {};
  for (int kk = 0; kk < Kpad; kk += 32) {
    // A frag: lane half 0 holds K[kk..kk+15]; half 1 holds K[kk+16..kk+31]
    v16bf af = ld_frag(arow + kk + half * 16, arow + kk + half * 16 + 8);
    v16bf bfi = ld_frag(Wti + wrow + kk + half * 16, Wti + wrow + kk + half * 16 + 8);
    accI = __builtin_amdgcn_wmma_f32_16x16x32_bf16(false, af, false, bfi,
                                                   (short)0, accI, false, false);
    v16bf bfc = ld_frag(Wtc + wrow + kk + half * 16, Wtc + wrow + kk + half * 16 + 8);
    accC = __builtin_amdgcn_wmma_f32_16x16x32_bf16(false, af, false, bfc,
                                                   (short)0, accC, false, false);
    v16bf bfo = ld_frag(Wto + wrow + kk + half * 16, Wto + wrow + kk + half * 16 + 8);
    accO = __builtin_amdgcn_wmma_f32_16x16x32_bf16(false, af, false, bfo,
                                                   (short)0, accO, false, false);
  }

  int col = nTile * 16 + l16;
  float vbi = bi[col], vbc = bc[col], vbo = bo[col], vw2 = wc2[col];
#pragma unroll
  for (int j = 0; j < 8; ++j) {
    int row = mTile * 16 + half * 8 + j;        // C/D layout: M = half*8 + j
    float Ig = sigmoidf_(accI[j] + vbi);
    float Tg = tanhf(accC[j] + vbc);
    float Cv = Ig * Tg;                          // C = F*0 + I*T
    float Og = sigmoidf_(accO[j] + vbo + vw2 * Cv);
    float h = Og * tanhf(Cv);
    out[(size_t)row * Fout + col] = h > 0.0f ? h : LRELU_SLOPE * h;
  }
}

// ------------------------- linear layer WMMA GEMM -------------------------
__global__ void __launch_bounds__(128)
k_lin_wmma(const __bf16* __restrict__ A, int Kd, const __bf16* __restrict__ Wt,
           const float* __restrict__ bias, float* out,
           int mTiles, int nTiles, int Fout, int doLrelu) {
  int wave = threadIdx.x >> 5;
  int lane = threadIdx.x & 31;
  int tile = blockIdx.x * 4 + wave;
  if (tile >= mTiles * nTiles) return;
  int nTile = tile % nTiles;
  int mTile = tile / nTiles;
  int half = lane >> 4;
  int l16 = lane & 15;

  const __bf16* arow = A + (size_t)(mTile * 16 + l16) * Kd;
  const __bf16* wr = Wt + (size_t)(nTile * 16 + l16) * Kd;

  v8f acc = {};
  for (int kk = 0; kk < Kd; kk += 32) {
    v16bf af = ld_frag(arow + kk + half * 16, arow + kk + half * 16 + 8);
    v16bf bf = ld_frag(wr + kk + half * 16, wr + kk + half * 16 + 8);
    acc = __builtin_amdgcn_wmma_f32_16x16x32_bf16(false, af, false, bf,
                                                  (short)0, acc, false, false);
  }
  int col = nTile * 16 + l16;
  float vb = bias[col];
#pragma unroll
  for (int j = 0; j < 8; ++j) {
    int row = mTile * 16 + half * 8 + j;
    float z = acc[j] + vb;
    if (doLrelu) z = z > 0.0f ? z : LRELU_SLOPE * z;
    out[(size_t)row * Fout + col] = z;
  }
}

// ------------------ final 64 -> 3 projection (scalar f32) -----------------
__global__ void k_lin3(const float* __restrict__ h, const float* __restrict__ W,
                       const float* __restrict__ b, float* out, int N) {
  long long i = blockIdx.x * (long long)blockDim.x + threadIdx.x;
  if (i >= (long long)N * 3) return;
  int r = (int)(i / 3), c = (int)(i % 3);
  float acc = b[c];
#pragma unroll
  for (int k = 0; k < 64; ++k) acc += h[(size_t)r * 64 + k] * W[k * 3 + c];
  out[i] = acc;
}

// --------------------------- host-side helpers ----------------------------
static void run_gconv_layer(const float* xin, int F, int Fout, int Kpad,
                            const float* Wx, const float* bx, const float* bh,
                            const float* wc, const float* bg,
                            const int* src, const int* dst, const float* wE,
                            int N, int E,
                            float* t1, float* t2, __bf16* Acat,
                            __bf16* wti, __bf16* wtc, __bf16* wto,
                            float* bi, float* bc, float* bo, float* wc2,
                            float* hout, hipStream_t stream) {
  const int TB = 256;
  long long nf = (long long)N * F;
  long long ekf = (long long)E * F;
  // T1 = Lhat @ x
  k_zero_f32<<<(int)((nf + TB - 1) / TB), TB, 0, stream>>>(t1, nf);
  k_spmm<<<(int)((ekf + TB - 1) / TB), TB, 0, stream>>>(xin, src, dst, wE, t1, F, 1.0f, ekf);
  // T2 = 2 * Lhat @ T1 - T0   (init to -T0, accumulate with scale 2)
  k_neg_copy<<<(int)((nf + TB - 1) / TB), TB, 0, stream>>>(xin, t2, nf);
  k_spmm<<<(int)((ekf + TB - 1) / TB), TB, 0, stream>>>(t1, src, dst, wE, t2, F, 2.0f, ekf);
  // pack [T0|T1|T2] -> bf16, K padded
  long long na = (long long)N * Kpad;
  k_pack3_bf16<<<(int)((na + TB - 1) / TB), TB, 0, stream>>>(xin, t1, t2, Acat, F, Kpad, na);
  // weights + biases
  int wt = 3 * Fout * Kpad;
  k_prep_cheb_w<<<(wt + TB - 1) / TB, TB, 0, stream>>>(Wx, wti, wtc, wto, F, Fout, Kpad, wt);
  k_prep_bias<<<(Fout + TB - 1) / TB, TB, 0, stream>>>(bx, bh, bg, wc, bi, bc, bo, wc2, Fout);
  // fused WMMA GEMM + LSTM cell + leaky ReLU
  int mTiles = N / 16, nTiles = Fout / 16;
  int tiles = mTiles * nTiles;
  k_cheb_lstm_wmma<<<(tiles + 3) / 4, 128, 0, stream>>>(Acat, Kpad, wti, wtc, wto,
                                                        bi, bc, bo, wc2, hout,
                                                        mTiles, nTiles, Fout);
}

static void run_lin(const float* hin, int Kd, int Fout, const float* W, const float* b,
                    __bf16* Abf, __bf16* Wt, float* hout, int N, int doLrelu,
                    hipStream_t stream) {
  const int TB = 256;
  long long n = (long long)N * Kd;
  k_pack_bf16<<<(int)((n + TB - 1) / TB), TB, 0, stream>>>(hin, Abf, n);
  int wt = Fout * Kd;
  k_prep_lin_w<<<(wt + TB - 1) / TB, TB, 0, stream>>>(W, Wt, Kd, Fout, wt);
  int mTiles = N / 16, nTiles = Fout / 16;
  int tiles = mTiles * nTiles;
  k_lin_wmma<<<(tiles + 3) / 4, 128, 0, stream>>>(Abf, Kd, Wt, b, hout,
                                                  mTiles, nTiles, Fout, doLrelu);
}

extern "C" void kernel_launch(void* const* d_in, const int* in_sizes, int n_in,
                              void* d_out, int out_size, void* d_ws, size_t ws_size,
                              hipStream_t stream) {
  const float* x = (const float*)d_in[0];
  const int* ei = (const int*)d_in[1];
  const int N = in_sizes[0] / 3;     // 50000 (multiple of 16)
  const int E = in_sizes[1] / 2;     // 800000
  const int* src = ei;
  const int* dst = ei + E;

  const float* l1_Wx = (const float*)d_in[2];
  const float* l1_bx = (const float*)d_in[3];
  const float* l1_bh = (const float*)d_in[5];   // Wh unused: H == 0
  const float* l1_wc = (const float*)d_in[6];
  const float* l1_b  = (const float*)d_in[7];
  const float* l2_Wx = (const float*)d_in[8];
  const float* l2_bx = (const float*)d_in[9];
  const float* l2_bh = (const float*)d_in[11];
  const float* l2_wc = (const float*)d_in[12];
  const float* l2_b  = (const float*)d_in[13];
  const float* l3_Wx = (const float*)d_in[14];
  const float* l3_bx = (const float*)d_in[15];
  const float* l3_bh = (const float*)d_in[17];
  const float* l3_wc = (const float*)d_in[18];
  const float* l3_b  = (const float*)d_in[19];
  const float* lin1_W = (const float*)d_in[20];
  const float* lin1_b = (const float*)d_in[21];
  const float* lin2_W = (const float*)d_in[22];
  const float* lin2_b = (const float*)d_in[23];
  const float* lin3_W = (const float*)d_in[24];
  const float* lin3_b = (const float*)d_in[25];

  // ---- workspace carve (256B aligned) ----
  char* ws = (char*)d_ws;
  size_t off = 0;
  auto carve = [&](size_t bytes) -> char* {
    char* p = ws + off;
    off += (bytes + 255) & ~(size_t)255;
    return p;
  };
  float* deg  = (float*)carve((size_t)N * 4);
  float* wE   = (float*)carve((size_t)E * 4);
  float* t1   = (float*)carve((size_t)N * 128 * 4);
  float* t2   = (float*)carve((size_t)N * 128 * 4);
  float* hA   = (float*)carve((size_t)N * 256 * 4);
  float* hB   = (float*)carve((size_t)N * 128 * 4);
  __bf16* Acat = (__bf16*)carve((size_t)N * 384 * 2);   // also reused for lin packs
  __bf16* wti = (__bf16*)carve((size_t)128 * 384 * 2);  // max gate weight (also lin Wt)
  __bf16* wtc = (__bf16*)carve((size_t)128 * 384 * 2);
  __bf16* wto = (__bf16*)carve((size_t)128 * 384 * 2);
  float* bi  = (float*)carve(256 * 4);
  float* bc  = (float*)carve(256 * 4);
  float* bo  = (float*)carve(256 * 4);
  float* wc2 = (float*)carve(256 * 4);

  const int TB = 256;
  // ---- edge normalization: w[e] = -rsqrt(deg[src]) * rsqrt(deg[dst]) ----
  k_zero_f32<<<(N + TB - 1) / TB, TB, 0, stream>>>(deg, N);
  k_degree<<<(E + TB - 1) / TB, TB, 0, stream>>>(src, deg, E);
  k_dis<<<(N + TB - 1) / TB, TB, 0, stream>>>(deg, N);
  k_edge_w<<<(E + TB - 1) / TB, TB, 0, stream>>>(src, dst, deg, wE, E);

  // ---- layer 1: Fin=3 -> 256 (3F=9 padded to K=32) ----
  run_gconv_layer(x, 3, 256, 32, l1_Wx, l1_bx, l1_bh, l1_wc, l1_b,
                  src, dst, wE, N, E, t1, t2, Acat, wti, wtc, wto,
                  bi, bc, bo, wc2, hA, stream);
  run_lin(hA, 256, 128, lin1_W, lin1_b, Acat, wti, hB, N, 1, stream);

  // ---- layer 2: 128 -> 128 (K=384) ----
  run_gconv_layer(hB, 128, 128, 384, l2_Wx, l2_bx, l2_bh, l2_wc, l2_b,
                  src, dst, wE, N, E, t1, t2, Acat, wti, wtc, wto,
                  bi, bc, bo, wc2, hA, stream);
  run_lin(hA, 128, 64, lin2_W, lin2_b, Acat, wti, hB, N, 1, stream);

  // ---- layer 3: 64 -> 64 (K=192) ----
  run_gconv_layer(hB, 64, 64, 192, l3_Wx, l3_bx, l3_bh, l3_wc, l3_b,
                  src, dst, wE, N, E, t1, t2, Acat, wti, wtc, wto,
                  bi, bc, bo, wc2, hA, stream);

  // ---- final projection 64 -> 3, full f32 ----
  long long n3 = (long long)N * 3;
  k_lin3<<<(int)((n3 + TB - 1) / TB), TB, 0, stream>>>(hA, lin3_W, lin3_b,
                                                       (float*)d_out, N);
}
